// CausalSelfAttention_84310208020548
// MI455X (gfx1250) — matrix operand order repaired and checked by reading, not compiled, not run on hardware
//
#include <hip/hip_runtime.h>
#include <hip/hip_bf16.h>

typedef __bf16 bf16_t;
typedef __attribute__((ext_vector_type(16))) __bf16 v16bf;
typedef __attribute__((ext_vector_type(8)))  __bf16 v8bf;
typedef __attribute__((ext_vector_type(8)))  float  v8f;

#define T_SEQ  4096
#define C_DIM  1024
#define H_NUM  16
#define D_HEAD 64

#define WMMA_BF16(a, b, c) \
  __builtin_amdgcn_wmma_f32_16x16x32_bf16(false, (a), false, (b), (short)0, (c), false, false)

__device__ __forceinline__ v16bf cat8(v8bf lo, v8bf hi) {
  return __builtin_shufflevector(lo, hi, 0, 1, 2, 3, 4, 5, 6, 7, 8, 9, 10, 11, 12, 13, 14, 15);
}

// A fragment (16x32 MxK) from row-major src (leading dim ld).
// lane&15 -> row (rowBase+m); K chunks {k0+8*half .. +8} and {k0+16+8*half .. +8}.
__device__ __forceinline__ v16bf load_a_frag(const bf16_t* __restrict__ src, int ld,
                                             int rowBase, int k0, int lane) {
  int m = lane & 15, half = lane >> 4;
  const bf16_t* p = src + (size_t)(rowBase + m) * ld + k0;
  v8bf lo = *(const v8bf*)(p + 8 * half);
  v8bf hi = *(const v8bf*)(p + 16 + 8 * half);
  return cat8(lo, hi);
}

// B fragment (32x16 KxN) for computing A * src^T: column n of B is row (rowBase+n)
// of row-major src; lane&15 -> n; K chunks {k0+8*half .. +8} and {k0+16+8*half .. +8}.
__device__ __forceinline__ v16bf load_bt_frag(const bf16_t* src, int ld,
                                              int rowBase, int k0, int lane) {
  int nn = lane & 15, half = lane >> 4;
  const bf16_t* p = src + (size_t)(rowBase + nn) * ld + k0 + 8 * half;
  v8bf lo = *(const v8bf*)(p);
  v8bf hi = *(const v8bf*)(p + 16);
  return cat8(lo, hi);
}

// One-lane async 16B copy global -> LDS (ASYNCcnt-tracked).
__device__ __forceinline__ void async_cp16(bf16_t* lds_dst, const bf16_t* gsrc) {
  unsigned lds_off = (unsigned)(unsigned long long)(void*)lds_dst;
  asm volatile("global_load_async_to_lds_b128 %0, %1, off"
               :: "v"(lds_off), "v"(gsrc)
               : "memory");
}

__device__ __forceinline__ void wait_async0() {
  asm volatile("s_wait_asynccnt 0x0" ::: "memory");
}

__global__ void cvt_f32_bf16(const float* __restrict__ in, bf16_t* __restrict__ out, int n) {
  int i = blockIdx.x * blockDim.x + threadIdx.x;
  int stride = gridDim.x * blockDim.x;
  for (; i < n; i += stride) out[i] = (bf16_t)in[i];
}

// qkv[t,f] = sum_c x[t,c] * w_attn[f,c]; split into Q,K (row-major bf16) and V transposed.
// 64x64 tile per wave (16 WMMAs per 8 fragment loads), 2x2 waves -> 128x128 per block.
__global__ void __launch_bounds__(128) gemm_qkv(const bf16_t* __restrict__ xb,
                                                const bf16_t* __restrict__ wb,
                                                bf16_t* __restrict__ Qb,
                                                bf16_t* __restrict__ Kb,
                                                bf16_t* __restrict__ Vt) {
  const int lane = threadIdx.x & 31;
  const int wave = threadIdx.x >> 5;
  const int nBlk = (3 * C_DIM) / 128;  // 24
  const int bm = blockIdx.x / nBlk, bn = blockIdx.x % nBlk;
  const int rowBase = bm * 128 + (wave >> 1) * 64;
  const int colBase = bn * 128 + (wave & 1) * 64;

  v8f acc[4][4] = {};
  for (int k0 = 0; k0 < C_DIM; k0 += 32) {
    v16bf a[4], b[4];
#pragma unroll
    for (int i = 0; i < 4; ++i) a[i] = load_a_frag(xb, C_DIM, rowBase + 16 * i, k0, lane);
#pragma unroll
    for (int j = 0; j < 4; ++j) b[j] = load_bt_frag(wb, C_DIM, colBase + 16 * j, k0, lane);
#pragma unroll
    for (int i = 0; i < 4; ++i)
#pragma unroll
      for (int j = 0; j < 4; ++j)
        acc[i][j] = WMMA_BF16(a[i], b[j], acc[i][j]);
  }

  const int n = lane & 15, mBase = (lane >> 4) * 8;
#pragma unroll
  for (int i = 0; i < 4; ++i) {
#pragma unroll
    for (int j = 0; j < 4; ++j) {
      const int f = colBase + j * 16 + n;
      for (int r = 0; r < 8; ++r) {
        int t = rowBase + i * 16 + mBase + r;
        bf16_t v = (bf16_t)acc[i][j][r];
        if (f < C_DIM)          Qb[(size_t)t * C_DIM + f] = v;
        else if (f < 2 * C_DIM) Kb[(size_t)t * C_DIM + (f - C_DIM)] = v;
        else                    Vt[(size_t)(f - 2 * C_DIM) * T_SEQ + t] = v;
      }
    }
  }
}

// Flash attention: block = 4 waves = 64 query rows of one head.
// K/V blocks staged into LDS via async-to-LDS, shared by all 4 waves.
__global__ void __launch_bounds__(128) attn_fwd(const bf16_t* __restrict__ Qb,
                                                const bf16_t* __restrict__ Kb,
                                                const bf16_t* __restrict__ Vt,
                                                bf16_t* __restrict__ yb) {
  __shared__ bf16_t ldsK[32 * 64];      // 32 keys x 64 d      (4 KB)
  __shared__ bf16_t ldsV[64 * 32];      // 64 d    x 32 keys   (4 KB)
  __shared__ bf16_t ldsP[4 * 16 * 32];  // per-wave P staging  (4 KB)
  const int tid  = threadIdx.x;
  const int lane = tid & 31;
  const int wave = tid >> 5;
  const int h    = blockIdx.x >> 6;   // 16 heads
  const int qblk = blockIdx.x & 63;   // 64 query-blocks of 64 rows
  const int qBase = qblk * 64 + wave * 16;
  const int n = lane & 15, half = lane >> 4, mBase = half * 8;
  bf16_t* P = ldsP + wave * (16 * 32);

  // Resident Q fragments: two 16x32 d-slices of this wave's 16 query rows.
  v16bf qf0 = load_a_frag(Qb, C_DIM, qBase, h * D_HEAD + 0, lane);
  v16bf qf1 = load_a_frag(Qb, C_DIM, qBase, h * D_HEAD + 32, lane);

  v8f y0 = {}, y1 = {}, y2 = {}, y3 = {};
  float rmax[8], rsum[8];
  for (int r = 0; r < 8; ++r) { rmax[r] = -1e30f; rsum[r] = 0.f; }

  const int kbEnd = qblk * 2 + 1;  // uniform across the block; extra blocks fully masked
  for (int kb = 0; kb <= kbEnd; ++kb) {
    __syncthreads();  // previous iteration's LDS readers done
    // Stage K tile: 32 rows x 64 d (8 chunks of 16B per row), 256 chunks / 128 threads.
    for (int c = tid; c < 256; c += 128) {
      int row = c >> 3, off8 = (c & 7) * 8;
      async_cp16(&ldsK[row * 64 + off8],
                 Kb + (size_t)(kb * 32 + row) * C_DIM + h * D_HEAD + off8);
    }
    // Stage V tile: 64 rows (d) x 32 keys (4 chunks per row), 256 chunks / 128 threads.
    for (int c = tid; c < 256; c += 128) {
      int row = c >> 2, off8 = (c & 3) * 8;
      async_cp16(&ldsV[row * 32 + off8],
                 Vt + (size_t)(h * D_HEAD + row) * T_SEQ + kb * 32 + off8);
    }
    wait_async0();
    __syncthreads();

    // S = Q K^T for 32 keys (two 16x16 C tiles), D=64 -> two K=32 WMMA steps each.
    v8f s0 = {}, s1 = {};
    s0 = WMMA_BF16(qf0, load_bt_frag(ldsK, D_HEAD, 0,  0,  lane), s0);
    s0 = WMMA_BF16(qf1, load_bt_frag(ldsK, D_HEAD, 0,  32, lane), s0);
    s1 = WMMA_BF16(qf0, load_bt_frag(ldsK, D_HEAD, 16, 0,  lane), s1);
    s1 = WMMA_BF16(qf1, load_bt_frag(ldsK, D_HEAD, 16, 32, lane), s1);

    const int key0 = kb * 32 + n, key1 = key0 + 16;
    float p0[8], p1[8];
    for (int r = 0; r < 8; ++r) {
      int q = qBase + mBase + r;
      float a0 = (key0 <= q) ? s0[r] * 0.125f : -1e30f;  // 1/sqrt(64)
      float a1 = (key1 <= q) ? s1[r] * 0.125f : -1e30f;
      float bm = fmaxf(a0, a1);
      for (int off = 1; off < 16; off <<= 1) bm = fmaxf(bm, __shfl_xor(bm, off, 32));
      float nm = fmaxf(rmax[r], bm);
      float corr = __expf(rmax[r] - nm);
      rmax[r] = nm;
      float e0 = __expf(a0 - nm), e1 = __expf(a1 - nm);
      p0[r] = e0; p1[r] = e1;
      float bs = e0 + e1;
      for (int off = 1; off < 16; off <<= 1) bs += __shfl_xor(bs, off, 32);
      rsum[r] = rsum[r] * corr + bs;
      y0[r] *= corr; y1[r] *= corr; y2[r] *= corr; y3[r] *= corr;
    }

    // Stage P (16x32, row-major bf16) through per-wave LDS: C-layout -> A-layout.
    for (int r = 0; r < 8; ++r) {
      P[(mBase + r) * 32 + n]      = (bf16_t)p0[r];
      P[(mBase + r) * 32 + 16 + n] = (bf16_t)p1[r];
    }
    asm volatile("s_wait_dscnt 0x0" ::: "memory");
    int m = lane & 15;
    v8bf plo = *(const v8bf*)(P + m * 32 + 8 * half);
    v8bf phi = *(const v8bf*)(P + m * 32 + 16 + 8 * half);
    v16bf pf = cat8(plo, phi);

    // y += P V, reading V fragments from LDS (rows = d, contiguous along keys).
    y0 = WMMA_BF16(pf, load_bt_frag(ldsV, 32, 0,  0, lane), y0);
    y1 = WMMA_BF16(pf, load_bt_frag(ldsV, 32, 16, 0, lane), y1);
    y2 = WMMA_BF16(pf, load_bt_frag(ldsV, 32, 32, 0, lane), y2);
    y3 = WMMA_BF16(pf, load_bt_frag(ldsV, 32, 48, 0, lane), y3);
  }

  for (int r = 0; r < 8; ++r) {
    float inv = 1.f / rsum[r];
    int t = qBase + mBase + r;
    bf16_t* yrow = yb + (size_t)t * C_DIM + h * D_HEAD;
    yrow[0 * 16 + n] = (bf16_t)(y0[r] * inv);
    yrow[1 * 16 + n] = (bf16_t)(y1[r] * inv);
    yrow[2 * 16 + n] = (bf16_t)(y2[r] * inv);
    yrow[3 * 16 + n] = (bf16_t)(y3[r] * inv);
  }
}

// out[t,o] = sum_c y[t,c] * w_proj[o,c]  (fp32 output), 64x64 tile per wave.
__global__ void __launch_bounds__(128) gemm_proj(const bf16_t* __restrict__ yb,
                                                 const bf16_t* __restrict__ wb,
                                                 float* __restrict__ out) {
  const int lane = threadIdx.x & 31;
  const int wave = threadIdx.x >> 5;
  const int nBlk = C_DIM / 128;  // 8
  const int bm = blockIdx.x / nBlk, bn = blockIdx.x % nBlk;
  const int rowBase = bm * 128 + (wave >> 1) * 64;
  const int colBase = bn * 128 + (wave & 1) * 64;

  v8f acc[4][4] = {};
  for (int k0 = 0; k0 < C_DIM; k0 += 32) {
    v16bf a[4], b[4];
#pragma unroll
    for (int i = 0; i < 4; ++i) a[i] = load_a_frag(yb, C_DIM, rowBase + 16 * i, k0, lane);
#pragma unroll
    for (int j = 0; j < 4; ++j) b[j] = load_bt_frag(wb, C_DIM, colBase + 16 * j, k0, lane);
#pragma unroll
    for (int i = 0; i < 4; ++i)
#pragma unroll
      for (int j = 0; j < 4; ++j)
        acc[i][j] = WMMA_BF16(a[i], b[j], acc[i][j]);
  }

  const int n = lane & 15, mBase = (lane >> 4) * 8;
#pragma unroll
  for (int i = 0; i < 4; ++i)
#pragma unroll
    for (int j = 0; j < 4; ++j)
      for (int r = 0; r < 8; ++r) {
        int t = rowBase + i * 16 + mBase + r;
        out[(size_t)t * C_DIM + colBase + j * 16 + n] = acc[i][j][r];
      }
}

extern "C" void kernel_launch(void* const* d_in, const int* in_sizes, int n_in,
                              void* d_out, int out_size, void* d_ws, size_t ws_size,
                              hipStream_t stream) {
  (void)in_sizes; (void)n_in; (void)out_size; (void)ws_size;
  const float* x      = (const float*)d_in[0];
  const float* w_attn = (const float*)d_in[1];
  const float* w_proj = (const float*)d_in[2];
  float* out = (float*)d_out;

  char* ws = (char*)d_ws;
  size_t off = 0;
  auto carve = [&](size_t bytes) -> void* {
    void* p = ws + off;
    off += (bytes + 255) & ~(size_t)255;
    return p;
  };
  const size_t TC = (size_t)T_SEQ * C_DIM;
  bf16_t* xb  = (bf16_t*)carve(TC * 2);
  bf16_t* wab = (bf16_t*)carve((size_t)3 * C_DIM * C_DIM * 2);
  bf16_t* wpb = (bf16_t*)carve((size_t)C_DIM * C_DIM * 2);
  bf16_t* Qb  = (bf16_t*)carve(TC * 2);
  bf16_t* Kb  = (bf16_t*)carve(TC * 2);
  bf16_t* Vt  = (bf16_t*)carve(TC * 2);
  bf16_t* yb  = (bf16_t*)carve(TC * 2);

  cvt_f32_bf16<<<2048, 256, 0, stream>>>(x, xb, (int)TC);
  cvt_f32_bf16<<<2048, 256, 0, stream>>>(w_attn, wab, 3 * C_DIM * C_DIM);
  cvt_f32_bf16<<<2048, 256, 0, stream>>>(w_proj, wpb, C_DIM * C_DIM);

  // (T/128)*(3C/128) = 32*24 = 768 blocks, each 128x128 output
  gemm_qkv<<<768, 128, 0, stream>>>(xb, wab, Qb, Kb, Vt);

  // 16 heads * 64 query-blocks of 64 rows
  attn_fwd<<<H_NUM * (T_SEQ / 64), 128, 0, stream>>>(Qb, Kb, Vt, yb);

  // (T/128)*(C/128) = 32*8 = 256 blocks
  gemm_proj<<<256, 128, 0, stream>>>(yb, wpb, out);
}
